// Attention_18107582120288
// MI455X (gfx1250) — matrix operand order repaired and checked by reading, not compiled, not run on hardware
//
#include <hip/hip_runtime.h>
#include <hip/hip_bf16.h>
#include <math.h>

#define HID    1024
#define SEQ    2048
#define NBATCH 32
#define KSTAGE 64
#define APAD   72   // padded LDS row stride in 16-bit elems (144B, 16B-aligned)
#define MBLK   128  // rows per block (k_scores)
#define NBLK   256  // cols per block (k_scores)
#define NCHUNK (HID / NBLK)   // 4 partial-score chunks
#define SCHUNK 16             // SEQ split for k_context
#define SSUB   (SEQ / SCHUNK) // 128 seq rows per context block

typedef __attribute__((ext_vector_type(16))) __bf16 v16bf;
typedef __attribute__((ext_vector_type(8)))  float  v8f;

union ABf { uint4 q[2]; v16bf v; };

__device__ __forceinline__ unsigned short f32_to_bf16_rne(float f) {
  unsigned int u = __float_as_uint(f);
  unsigned int r = u + 0x7FFFu + ((u >> 16) & 1u);
  return (unsigned short)(r >> 16);
}

__device__ __forceinline__ float half16_sum(float v) {
  v += __shfl_xor(v, 1);
  v += __shfl_xor(v, 2);
  v += __shfl_xor(v, 4);
  v += __shfl_xor(v, 8);
  return v;
}

// ---------------------------------------------------------------------------
// Prep: w_eT[n][e] = bf16(attn_w[1024+e][n])  (LDS tile transpose)
// ---------------------------------------------------------------------------
__global__ void k_prep_weT(const float* __restrict__ attn_w,
                           unsigned short* __restrict__ weT) {
  __shared__ unsigned short tile[32][33];
  const int tx = threadIdx.x;          // 0..31
  const int ty = threadIdx.y;          // 0..7
  const int n0 = blockIdx.x * 32;
  const int e0 = blockIdx.y * 32;
#pragma unroll
  for (int j = 0; j < 4; ++j) {
    int e = e0 + ty + j * 8;
    tile[ty + j * 8][tx] = f32_to_bf16_rne(attn_w[(size_t)(HID + e) * HID + n0 + tx]);
  }
  __syncthreads();
#pragma unroll
  for (int j = 0; j < 4; ++j) {
    int n = n0 + ty + j * 8;
    weT[(size_t)n * HID + e0 + tx] = tile[tx][ty + j * 8];
  }
}

// ---------------------------------------------------------------------------
// h_proj + bias: hb[b][k] = sum_h hidden[b][h] * attn_w[h][k] + attn_b[k]
// ---------------------------------------------------------------------------
__global__ void k_hproj(const float* __restrict__ hidden,
                        const float* __restrict__ attn_w,
                        const float* __restrict__ attn_b,
                        float* __restrict__ hb) {
  __shared__ float hid[HID];
  const int b = blockIdx.y;
  const int k = blockIdx.x * 256 + threadIdx.x;
  for (int i = threadIdx.x; i < HID; i += 256) hid[i] = hidden[(size_t)b * HID + i];
  __syncthreads();
  float acc = 0.f;
#pragma unroll 8
  for (int h = 0; h < HID; ++h) acc += hid[h] * attn_w[(size_t)h * HID + k];
  hb[(size_t)b * HID + k] = acc + attn_b[k];
}

// ---------------------------------------------------------------------------
// Fused GEMM + tanh + partial v-dot.
// Block tile: M=128 (4 wave-groups x 32 rows), N=256 (2 wave-groups x 128 cols),
// K=1024.  A and B staged through LDS; B register double-buffered so LDS
// latency overlaps WMMA issue.  Deterministic partial scores, one writer/slot.
// ---------------------------------------------------------------------------
__launch_bounds__(256)
__global__ void k_scores(const float* __restrict__ enc,
                         const unsigned short* __restrict__ weT,
                         const float* __restrict__ hb,
                         const float* __restrict__ v_w,
                         float* __restrict__ attP) {
  __shared__ __align__(16) unsigned short As[MBLK * APAD];   // 18 KB
  __shared__ __align__(16) unsigned short Bs[NBLK * APAD];   // 36 KB
  __shared__ float attL[2][MBLK];                            // per-N-group partials

  const int b    = blockIdx.z;
  const int m0   = blockIdx.x * MBLK;
  const int n0   = blockIdx.y * NBLK;
  const int tid  = threadIdx.x;
  const int lane = tid & 31;
  const int wave = tid >> 5;
  const int wm   = wave & 3;       // M-group: rows wm*32 .. wm*32+31
  const int wn   = wave >> 2;      // N-group: cols wn*128 .. wn*128+127
  const int hi   = lane >> 4;
  const int l15  = lane & 15;

  v8f zero = {0.f, 0.f, 0.f, 0.f, 0.f, 0.f, 0.f, 0.f};
  v8f acc[2][8];
#pragma unroll
  for (int mt = 0; mt < 2; ++mt)
#pragma unroll
    for (int t = 0; t < 8; ++t) acc[mt][t] = zero;

  const float* aBase = enc + ((size_t)b * SEQ + m0) * HID;
  const int srow = tid >> 3;        // 0..31
  const int scol = (tid & 7) * 8;   // 0..56 (halves / floats)

  auto loadB = [&](ABf& B, int t, int kc) {
    const int offB = (wn * 128 + t * 16 + l15) * APAD + kc + hi * 8;
    B.q[0] = *(const uint4*)&Bs[offB];
    B.q[1] = *(const uint4*)&Bs[offB + 16];
  };

  for (int k0 = 0; k0 < HID; k0 += KSTAGE) {
    // ---- stage A: 128 x 64 f32 -> bf16 (4 rows / thread) ----
#pragma unroll
    for (int r = 0; r < 4; ++r) {
      const int row = srow + r * 32;
      const float* src = aBase + (size_t)row * HID + k0 + scol;
      float4 f0 = *(const float4*)(src);
      float4 f1 = *(const float4*)(src + 4);
      uint4 p;
      p.x = (unsigned)f32_to_bf16_rne(f0.x) | ((unsigned)f32_to_bf16_rne(f0.y) << 16);
      p.y = (unsigned)f32_to_bf16_rne(f0.z) | ((unsigned)f32_to_bf16_rne(f0.w) << 16);
      p.z = (unsigned)f32_to_bf16_rne(f1.x) | ((unsigned)f32_to_bf16_rne(f1.y) << 16);
      p.w = (unsigned)f32_to_bf16_rne(f1.z) | ((unsigned)f32_to_bf16_rne(f1.w) << 16);
      *(uint4*)&As[row * APAD + scol] = p;
    }
    // ---- stage B: 256 cols x 64 halves, straight copy (8 rows / thread) ----
#pragma unroll
    for (int r = 0; r < 8; ++r) {
      const int n = srow + r * 32;
      const uint4 q = *(const uint4*)(weT + (size_t)(n0 + n) * HID + k0 + scol);
      *(uint4*)&Bs[n * APAD + scol] = q;
    }
    __syncthreads();

#pragma unroll
    for (int kc = 0; kc < KSTAGE; kc += 32) {
      ABf A0, A1;
      const int offA = (wm * 32 + l15) * APAD + kc + hi * 8;
      A0.q[0] = *(const uint4*)&As[offA];
      A0.q[1] = *(const uint4*)&As[offA + 16];
      A1.q[0] = *(const uint4*)&As[offA + 16 * APAD];
      A1.q[1] = *(const uint4*)&As[offA + 16 * APAD + 16];

      ABf Bbuf[2];
      loadB(Bbuf[0], 0, kc);
#pragma unroll
      for (int t = 0; t < 8; ++t) {
        if (t < 7) loadB(Bbuf[(t + 1) & 1], t + 1, kc);   // prefetch next tile
        const ABf& Bc = Bbuf[t & 1];
        acc[0][t] = __builtin_amdgcn_wmma_f32_16x16x32_bf16(
            false, A0.v, false, Bc.v, (short)0, acc[0][t], false, false);
        acc[1][t] = __builtin_amdgcn_wmma_f32_16x16x32_bf16(
            false, A1.v, false, Bc.v, (short)0, acc[1][t], false, false);
      }
    }
    __syncthreads();
  }

  // ---- epilogue: tanh(acc + hb) . v_w over this block's 256 columns ----
  float hbv[8], vwv[8];
#pragma unroll
  for (int t = 0; t < 8; ++t) {
    int c = n0 + wn * 128 + t * 16 + l15;
    hbv[t] = hb[(size_t)b * HID + c];
    vwv[t] = v_w[c];
  }
#pragma unroll
  for (int mt = 0; mt < 2; ++mt) {
    float part[8];
#pragma unroll
    for (int i = 0; i < 8; ++i) part[i] = 0.f;
#pragma unroll
    for (int t = 0; t < 8; ++t)
#pragma unroll
      for (int i = 0; i < 8; ++i)
        part[i] += tanhf(acc[mt][t][i] + hbv[t]) * vwv[t];
#pragma unroll
    for (int i = 0; i < 8; ++i) {
      float v = half16_sum(part[i]);   // sum over 16 lane-columns
      if (l15 == 0) {
        // C/D layout: row = i + 8*hi within tile; unique (wn,row) writer
        attL[wn][wm * 32 + mt * 16 + i + 8 * hi] = v;
      }
    }
  }
  __syncthreads();
  if (tid < MBLK) {
    // fixed-order combine of the two N-groups -> deterministic
    attP[((size_t)blockIdx.y * NBATCH + b) * SEQ + m0 + tid] =
        attL[0][tid] + attL[1][tid];
  }
}

// ---------------------------------------------------------------------------
// Combine 4 partial-score chunks (fixed order) + softmax over SEQ per batch
// ---------------------------------------------------------------------------
__global__ void k_softmax(const float* __restrict__ attP,
                          float* __restrict__ wgt) {
  __shared__ float red[256];
  const int b = blockIdx.x;
  const int tid = threadIdx.x;
  float v[8];
  float mx = -INFINITY;
#pragma unroll
  for (int j = 0; j < 8; ++j) {
    const int s = tid + j * 256;
    float a = 0.f;
#pragma unroll
    for (int c = 0; c < NCHUNK; ++c)
      a += attP[((size_t)c * NBATCH + b) * SEQ + s];
    v[j] = a;
    mx = fmaxf(mx, a);
  }
  red[tid] = mx;
  __syncthreads();
  for (int s = 128; s > 0; s >>= 1) {
    if (tid < s) red[tid] = fmaxf(red[tid], red[tid + s]);
    __syncthreads();
  }
  mx = red[0];
  __syncthreads();
  float sum = 0.f;
#pragma unroll
  for (int j = 0; j < 8; ++j) {
    v[j] = __expf(v[j] - mx);
    sum += v[j];
  }
  red[tid] = sum;
  __syncthreads();
  for (int s = 128; s > 0; s >>= 1) {
    if (tid < s) red[tid] += red[tid + s];
    __syncthreads();
  }
  const float inv = 1.f / red[0];
#pragma unroll
  for (int j = 0; j < 8; ++j)
    wgt[(size_t)b * SEQ + tid + j * 256] = v[j] * inv;
}

// ---------------------------------------------------------------------------
// Partial context: ctxP[c][b][e] = sum_{s in chunk c} wgt[b][s]*enc[b][s][e]
// SEQ split 16x for memory-level parallelism (HBM roofline: need many MB
// of loads in flight to reach 23.3 TB/s).
// ---------------------------------------------------------------------------
__global__ void k_context(const float* __restrict__ wgt,
                          const float* __restrict__ enc,
                          float* __restrict__ ctxP) {
  const int c = blockIdx.y;             // seq chunk
  const int b = blockIdx.z;
  const int e = blockIdx.x * 256 + threadIdx.x;
  const int s0 = c * SSUB;
  const float* w  = wgt + (size_t)b * SEQ + s0;
  const float* ep = enc + ((size_t)b * SEQ + s0) * HID + e;
  float acc = 0.f;
#pragma unroll 8
  for (int s = 0; s < SSUB; ++s) acc += w[s] * ep[(size_t)s * HID];
  ctxP[((size_t)c * NBATCH + b) * HID + e] = acc;
}

// ---------------------------------------------------------------------------
// Fixed-order reduction of the 16 context partials -> out (deterministic)
// ---------------------------------------------------------------------------
__global__ void k_ctxsum(const float* __restrict__ ctxP,
                         float* __restrict__ out) {
  const int i = blockIdx.x * 256 + threadIdx.x;   // 0 .. 32*1024-1
  float a = 0.f;
#pragma unroll
  for (int c = 0; c < SCHUNK; ++c)
    a += ctxP[(size_t)c * NBATCH * HID + i];
  out[i] = a;
}

// ---------------------------------------------------------------------------
extern "C" void kernel_launch(void* const* d_in, const int* in_sizes, int n_in,
                              void* d_out, int out_size, void* d_ws, size_t ws_size,
                              hipStream_t stream) {
  const float* hidden = (const float*)d_in[0];   // (32,1,1024)
  const float* enc    = (const float*)d_in[1];   // (32,2048,1024)
  const float* attn_w = (const float*)d_in[2];   // (2048,1024)
  const float* attn_b = (const float*)d_in[3];   // (1024,)
  const float* v_w    = (const float*)d_in[4];   // (1024,)
  float* out = (float*)d_out;                    // (32,1,1024)

  char* ws = (char*)d_ws;
  size_t off = 0;
  float* hb   = (float*)(ws + off); off += (size_t)NBATCH * HID * 4;          // 128 KB
  float* attP = (float*)(ws + off); off += (size_t)NCHUNK * NBATCH * SEQ * 4; // 1 MB
  float* wgt  = (float*)(ws + off); off += (size_t)NBATCH * SEQ * 4;          // 256 KB
  float* ctxP = (float*)(ws + off); off += (size_t)SCHUNK * NBATCH * HID * 4; // 2 MB
  unsigned short* weT = (unsigned short*)(ws + off);                          // 2 MB

  k_prep_weT<<<dim3(32, 32), dim3(32, 8), 0, stream>>>(attn_w, weT);
  k_hproj  <<<dim3(4, NBATCH), 256, 0, stream>>>(hidden, attn_w, attn_b, hb);
  k_scores <<<dim3(SEQ / MBLK, NCHUNK, NBATCH), 256, 0, stream>>>(enc, weT, hb, v_w, attP);
  k_softmax<<<NBATCH, 256, 0, stream>>>(attP, wgt);
  k_context<<<dim3(HID / 256, SCHUNK, NBATCH), 256, 0, stream>>>(wgt, enc, ctxP);
  k_ctxsum <<<dim3(NBATCH * HID / 256), 256, 0, stream>>>(ctxP, out);
}
